// ActorCriticMoE_37941741092911
// MI455X (gfx1250) — matrix operand order repaired and checked by reading, not compiled, not run on hardware
//
#include <hip/hip_runtime.h>

// ---------------------------------------------------------------------------
// ActorCriticMoE for MI455X (gfx1250, wave32, WMMA bf16 16x16x32)
// B=32768, OBS=256, ACT=12, H=256, K=8 experts, GH=128, TOPK=2
// Software-pipelined: async global->LDS weight staging double-buffered in
// 64KB column-halves, overlapped with WMMA compute; B-fragment DS loads
// explicitly pipelined 2 tiles ahead of the WMMA consuming them.
// ---------------------------------------------------------------------------

#define B_   32768
#define OBS_ 256
#define ACT_ 12
#define H_   256
#define K_   8
#define GH_  128

#define BM   128                  // batch rows per block (8 waves x 16-row strip)
#define WSTRIDE   (H_ * H_)       // 65536 halves = 128 KB per expert layer
#define WHALF     (128 * H_)      // 32768 halves = 64 KB half (128 output cols)
#define W4STRIDE  (16 * H_)       // 4096 halves (ACT padded 12->16)

typedef __attribute__((ext_vector_type(16))) __bf16 bf16x16;
typedef __attribute__((ext_vector_type(8)))  float  f32x8;
typedef __attribute__((ext_vector_type(4)))  int    v4i;

union Frag {            // one WMMA bf16 operand: 16 halves / lane = 8 VGPRs
    bf16x16 v;
    uint4   q[2];
};

// ---- gfx1250 async global->LDS copy (ASYNCcnt-tracked), with fallback -----
#if defined(__has_builtin)
#if __has_builtin(__builtin_amdgcn_global_load_async_to_lds_b128)
#define HAVE_ASYNC_LDS 1
#endif
#endif

__device__ __forceinline__ void async_copy16(const void* g, void* l) {
#ifdef HAVE_ASYNC_LDS
    __builtin_amdgcn_global_load_async_to_lds_b128(
        (__attribute__((address_space(1))) v4i*)g,
        (__attribute__((address_space(3))) v4i*)l, 0, 0);
#else
    *(uint4*)l = *(const uint4*)g;
#endif
}

#if defined(__has_builtin) && __has_builtin(__builtin_amdgcn_s_wait_asynccnt)
#define WAIT_ASYNC(n) __builtin_amdgcn_s_wait_asynccnt(n)
#else
#define WAIT_ASYNC(n) asm volatile("s_wait_asynccnt %0" :: "i"(n) : "memory")
#endif

// float -> bf16 round-nearest-even (activations: no NaN path needed)
__device__ __forceinline__ unsigned short f2bf(float f) {
    union { float f; unsigned u; } c; c.f = f;
    unsigned u = c.u;
    return (unsigned short)((u + 0x7FFFu + ((u >> 16) & 1u)) >> 16);
}

__device__ __forceinline__ float eluf(float v) {
    return v > 0.f ? v : (__expf(v) - 1.f);
}

// ---------------------------------------------------------------------------
// Weight convert + transpose: W[K][IN][OUT] f32  ->  Wt[K][OUTP][IN] bf16
// ---------------------------------------------------------------------------
__global__ void moe_convw_kernel(const float* __restrict__ W,
                                 unsigned short* __restrict__ Wt,
                                 int IN, int OUT, int OUTP, int total) {
    int gid = blockIdx.x * blockDim.x + threadIdx.x;
    if (gid >= total) return;
    int per = OUTP * IN;
    int k   = gid / per;
    int rem = gid - k * per;
    int n   = rem / IN;
    int h   = rem - n * IN;
    float v = (n < OUT) ? W[((size_t)k * IN + h) * OUT + n] : 0.f;
    Wt[gid] = f2bf(v);
}

// ---------------------------------------------------------------------------
// Gate: elu(x@Wg1+bg1)@Wg2+bg2 -> top-2 masked softmax -> gateW[B][K] f32
// ---------------------------------------------------------------------------
#define GBR 32
__global__ void moe_gate_kernel(const float* __restrict__ x,
                                const float* __restrict__ Wg1,
                                const float* __restrict__ bg1,
                                const float* __restrict__ Wg2,
                                const float* __restrict__ bg2,
                                float* __restrict__ gateW) {
    __shared__ float sx[GBR * OBS_];
    __shared__ float shg[GBR * GH_];
    __shared__ float slog[GBR * K_];
    const int tid = threadIdx.x;
    const size_t base = (size_t)blockIdx.x * GBR * OBS_;

    const float4* xs = (const float4*)(x + base);
    for (int j = tid; j < GBR * OBS_ / 4; j += 256)
        ((float4*)sx)[j] = xs[j];
    __syncthreads();

    for (int i = 0; i < 16; ++i) {
        int o = tid + i * 256;
        int row = o >> 7, g = o & (GH_ - 1);
        float s = bg1[g];
        const float* xr = sx + row * OBS_;
        for (int c = 0; c < OBS_; ++c)
            s += xr[c] * Wg1[c * GH_ + g];
        shg[row * GH_ + g] = eluf(s);
    }
    __syncthreads();

    {
        int row = tid >> 3, kk = tid & 7;
        float s = bg2[kk];
        const float* hr = shg + row * GH_;
        for (int g = 0; g < GH_; ++g)
            s += hr[g] * Wg2[g * K_ + kk];
        slog[row * K_ + kk] = s;
    }
    __syncthreads();

    if (tid < GBR) {
        float l[K_];
        for (int i = 0; i < K_; ++i) l[i] = slog[tid * K_ + i];
        int i1 = 0;
        for (int i = 1; i < K_; ++i) if (l[i] > l[i1]) i1 = i;
        int i2 = (i1 == 0) ? 1 : 0;
        for (int i = 0; i < K_; ++i) { if (i == i1) continue; if (l[i] > l[i2]) i2 = i; }
        float e2 = __expf(l[i2] - l[i1]);
        float d  = 1.f + e2;
        size_t gr = base / OBS_ + tid;
        for (int i = 0; i < K_; ++i)
            gateW[gr * K_ + i] = (i == i1) ? (1.f / d) : ((i == i2) ? (e2 / d) : 0.f);
    }
}

// ---------------------------------------------------------------------------
// Fragment loaders
// ---------------------------------------------------------------------------
__device__ __forceinline__ void load_afrags(Frag (&a)[8], const unsigned short* inB,
                                            int stripBase, int lane, int hiSel) {
    const int lh = lane & 15;
    const int arow = (stripBase + lh) * H_;
    #pragma unroll
    for (int kb = 0; kb < 8; ++kb) {                 // A: 16x32 tiles, K=0..255
        int bp = arow + kb * 32 + hiSel * 8;         // lanes>=16 hold K+8
        a[kb].q[0] = *(const uint4*)(inB + bp);
        a[kb].q[1] = *(const uint4*)(inB + bp + 16);
    }
}

__device__ __forceinline__ void loadB(Frag& b, const unsigned short* wB,
                                      int nrow, int kb, int hiSel) {
    int kp = nrow + kb * 32 + hiSel * 8;             // lanes>=16 hold K+8
    b.q[0] = *(const uint4*)(wB + kp);
    b.q[1] = *(const uint4*)(wB + kp + 16);
}

// ---------------------------------------------------------------------------
// Compute 128 output columns (8 N-tiles) from one half-weight LDS buffer.
// B-fragment loads run 2 tiles ahead of the consuming WMMA (rotating 3-slot
// buffer, all indices static under full kb unroll). Per-tile bias load is
// issued before the 8-WMMA chain and consumed only in the epilogue.
// ---------------------------------------------------------------------------
__device__ __forceinline__ void compute_half(const Frag (&a)[8],
                                             const unsigned short* wB,
                                             unsigned short* outB,
                                             const float* __restrict__ bias,
                                             int colBase, int stripBase,
                                             int lane, int hiSel) {
    const int lh = lane & 15;
    #pragma unroll 2
    for (int jt = 0; jt < 8; ++jt) {
        const int c   = colBase + jt * 16 + lh;
        const float bbv = bias[c];                   // latency hidden by WMMAs
        const int nrow = (jt * 16 + lh) * H_;        // row inside this half buf
        Frag bb[3];
        loadB(bb[0], wB, nrow, 0, hiSel);
        loadB(bb[1], wB, nrow, 1, hiSel);
        f32x8 acc = {0.f, 0.f, 0.f, 0.f, 0.f, 0.f, 0.f, 0.f};
        #pragma unroll
        for (int kb = 0; kb < 8; ++kb) {
            if (kb + 2 < 8) loadB(bb[(kb + 2) % 3], wB, nrow, kb + 2, hiSel);
            acc = __builtin_amdgcn_wmma_f32_16x16x32_bf16(
                false, a[kb].v, false, bb[kb % 3].v, (short)0, acc, false, false);
        }
        #pragma unroll
        for (int r = 0; r < 8; ++r) {                // C: VGPR r -> M=r / r+8
            int Mr = hiSel ? (r + 8) : r;
            float v = eluf(acc[r] + bbv);
            outB[(stripBase + Mr) * H_ + c] = f2bf(v);
        }
    }
}

// 256 -> 16 (ACT padded) output layer, returns raw f32 accumulator tile
__device__ __forceinline__ f32x8 layer_out16(const unsigned short* inB,
                                             const unsigned short* wB,
                                             int stripBase, int lane, int hiSel) {
    const int lh = lane & 15;
    const int nrow = lh * H_;
    Frag a[8];
    load_afrags(a, inB, stripBase, lane, hiSel);
    Frag bb[3];
    loadB(bb[0], wB, nrow, 0, hiSel);
    loadB(bb[1], wB, nrow, 1, hiSel);
    f32x8 acc = {0.f, 0.f, 0.f, 0.f, 0.f, 0.f, 0.f, 0.f};
    #pragma unroll
    for (int kb = 0; kb < 8; ++kb) {
        if (kb + 2 < 8) loadB(bb[(kb + 2) % 3], wB, nrow, kb + 2, hiSel);
        acc = __builtin_amdgcn_wmma_f32_16x16x32_bf16(
            false, a[kb].v, false, bb[kb % 3].v, (short)0, acc, false, false);
    }
    return acc;
}

// async-issue one 64KB weight half (16 b128 ops per thread)
__device__ __forceinline__ void issue_half(const unsigned short* src,
                                           unsigned short* dst, int tid) {
    const uint4* s = (const uint4*)src;
    uint4* d = (uint4*)dst;
    #pragma unroll
    for (int i = 0; i < 16; ++i)
        async_copy16(s + tid + (i << 8), d + tid + (i << 8));
}

// async-issue the 8KB W4 block (2 b128 ops per thread)
__device__ __forceinline__ void issue_w4(const unsigned short* src,
                                         unsigned short* dst, int tid) {
    const uint4* s = (const uint4*)src;
    uint4* d = (uint4*)dst;
    async_copy16(s + tid, d + tid);
    async_copy16(s + tid + 256, d + tid + 256);
}

// ---------------------------------------------------------------------------
// Fused MoE expert kernel, stage-pipelined:
// 7 stages/expert (L1h0 L1h1 L2h0 L2h1 L3h0 L3h1 L4); at each stage we
// compute from buf[p] while the next stage's weights are in flight, then
// wait ASYNCcnt, barrier, and issue the stage+2 load into the freed buffer.
// Activations are strip-private per wave -> no barriers on ACT ever.
// ---------------------------------------------------------------------------
__global__ __launch_bounds__(256) void moe_main_kernel(
        const float* __restrict__ x,
        const unsigned short* __restrict__ W1t,
        const unsigned short* __restrict__ W2t,
        const unsigned short* __restrict__ W3t,
        const unsigned short* __restrict__ W4t,
        const float* __restrict__ b1, const float* __restrict__ b2,
        const float* __restrict__ b3, const float* __restrict__ b4,
        const float* __restrict__ gateW,
        float* __restrict__ out) {
    extern __shared__ unsigned char smemRaw[];
    unsigned short* XT   = (unsigned short*)smemRaw;      // [BM][256] bf16 x
    unsigned short* ACTb = XT + BM * H_;                  // [BM][256] bf16 act
    unsigned short* WB0  = ACTb + BM * H_;                // 64 KB weight buf 0
    unsigned short* WB1  = WB0 + WHALF;                   // 64 KB weight buf 1
    float* sgw = (float*)(WB1 + WHALF);                   // [BM][8] gate tile

    const int tid   = threadIdx.x;
    const int lane  = tid & 31;
    const int hiSel = (lane >> 4) & 1;
    const int stripBase = (tid >> 5) * 16;                // wave's 16-row strip
    const int rb = blockIdx.x * BM;

    // x tile -> bf16 in LDS (read once; reused by all 8 experts) + gate tile
    {
        const float4* xs = (const float4*)(x + (size_t)rb * OBS_);
        for (int j = tid; j < BM * OBS_ / 4; j += 256) {
            float4 v = xs[j];
            ushort4 h;
            h.x = f2bf(v.x); h.y = f2bf(v.y); h.z = f2bf(v.z); h.w = f2bf(v.w);
            ((ushort4*)XT)[j] = h;
        }
        for (int j = tid; j < BM * K_; j += 256)
            sgw[j] = gateW[(size_t)rb * K_ + j];
    }

    f32x8 outAcc = {0.f, 0.f, 0.f, 0.f, 0.f, 0.f, 0.f, 0.f};

    // pipeline prologue: stage0 (L1h0 of k=0) -> WB0, stage1 (L1h1) -> WB1
    issue_half(W1t, WB0, tid);
    issue_half(W1t + WHALF, WB1, tid);
    WAIT_ASYNC(16);                 // half0 complete (async loads retire in order)
    __syncthreads();                // all waves' half0 portions + XT/sgw visible
    int p = 0;

    Frag a[8];
    for (int k = 0; k < K_; ++k) {
        const unsigned short* w2 = W2t + (size_t)k * WSTRIDE;
        const unsigned short* w3 = W3t + (size_t)k * WSTRIDE;
        const unsigned short* w4 = W4t + (size_t)k * W4STRIDE;
        const unsigned short* n1 = (k + 1 < K_) ? W1t + (size_t)(k + 1) * WSTRIDE
                                                : (const unsigned short*)0;
        const float* bb1 = b1 + k * H_;
        const float* bb2 = b2 + k * H_;
        const float* bb3 = b3 + k * H_;

        // --- L1 (input = XT) ---
        load_afrags(a, XT, stripBase, lane, hiSel);
        compute_half(a, p ? WB1 : WB0, ACTb, bb1, 0, stripBase, lane, hiSel);
        WAIT_ASYNC(0); __syncthreads();
        issue_half(w2, p ? WB1 : WB0, tid);                 // L2h0
        p ^= 1;
        compute_half(a, p ? WB1 : WB0, ACTb, bb1, 128, stripBase, lane, hiSel);
        WAIT_ASYNC(0); __syncthreads();
        issue_half(w2 + WHALF, p ? WB1 : WB0, tid);         // L2h1
        p ^= 1;

        // --- L2 (in-place on ACT; strip A-frags register-resident first) ---
        load_afrags(a, ACTb, stripBase, lane, hiSel);
        compute_half(a, p ? WB1 : WB0, ACTb, bb2, 0, stripBase, lane, hiSel);
        WAIT_ASYNC(0); __syncthreads();
        issue_half(w3, p ? WB1 : WB0, tid);                 // L3h0
        p ^= 1;
        compute_half(a, p ? WB1 : WB0, ACTb, bb2, 128, stripBase, lane, hiSel);
        WAIT_ASYNC(0); __syncthreads();
        issue_half(w3 + WHALF, p ? WB1 : WB0, tid);         // L3h1
        p ^= 1;

        // --- L3 ---
        load_afrags(a, ACTb, stripBase, lane, hiSel);
        compute_half(a, p ? WB1 : WB0, ACTb, bb3, 0, stripBase, lane, hiSel);
        WAIT_ASYNC(0); __syncthreads();
        issue_w4(w4, p ? WB1 : WB0, tid);                   // L4
        p ^= 1;
        compute_half(a, p ? WB1 : WB0, ACTb, bb3, 128, stripBase, lane, hiSel);
        WAIT_ASYNC(0); __syncthreads();
        if (n1) issue_half(n1, p ? WB1 : WB0, tid);         // next expert L1h0
        p ^= 1;

        // --- L4 + gated combine ---
        f32x8 e = layer_out16(ACTb, p ? WB1 : WB0, stripBase, lane, hiSel);
        {
            const int c = lane & 15;
            const float bc = (c < ACT_) ? b4[k * ACT_ + c] : 0.f;
            #pragma unroll
            for (int r = 0; r < 8; ++r) {
                int Mr = hiSel ? (r + 8) : r;
                float w = sgw[(stripBase + Mr) * K_ + k];
                outAcc[r] += w * (e[r] + bc);
            }
        }
        WAIT_ASYNC(0); __syncthreads();
        if (n1) issue_half(n1 + WHALF, p ? WB1 : WB0, tid); // next expert L1h1
        p ^= 1;
    }

    // write out[B,12]
    const int c = lane & 15;
    if (c < ACT_) {
        #pragma unroll
        for (int r = 0; r < 8; ++r) {
            int Mr = hiSel ? (r + 8) : r;
            out[(size_t)(rb + stripBase + Mr) * ACT_ + c] = outAcc[r];
        }
    }
}

// ---------------------------------------------------------------------------
extern "C" void kernel_launch(void* const* d_in, const int* in_sizes, int n_in,
                              void* d_out, int out_size, void* d_ws, size_t ws_size,
                              hipStream_t stream) {
    (void)in_sizes; (void)n_in; (void)out_size; (void)ws_size;
    const float* x   = (const float*)d_in[0];
    const float* W1  = (const float*)d_in[1];
    const float* b1  = (const float*)d_in[2];
    const float* W2  = (const float*)d_in[3];
    const float* b2  = (const float*)d_in[4];
    const float* W3  = (const float*)d_in[5];
    const float* b3  = (const float*)d_in[6];
    const float* W4  = (const float*)d_in[7];
    const float* b4  = (const float*)d_in[8];
    const float* Wg1 = (const float*)d_in[9];
    const float* bg1 = (const float*)d_in[10];
    const float* Wg2 = (const float*)d_in[11];
    const float* bg2 = (const float*)d_in[12];
    float* out = (float*)d_out;

    unsigned char* ws = (unsigned char*)d_ws;
    float* gateW = (float*)ws;                                   // 1 MB
    unsigned short* W1t = (unsigned short*)(ws + (1u << 20));    // 1 MB
    unsigned short* W2t = (unsigned short*)(ws + (2u << 20));    // 1 MB
    unsigned short* W3t = (unsigned short*)(ws + (3u << 20));    // 1 MB
    unsigned short* W4t = (unsigned short*)(ws + (4u << 20));    // 64 KB

    {
        int tot = K_ * H_ * H_;
        moe_convw_kernel<<<(tot + 255) / 256, 256, 0, stream>>>(W1, W1t, OBS_, H_, H_, tot);
        moe_convw_kernel<<<(tot + 255) / 256, 256, 0, stream>>>(W2, W2t, H_, H_, H_, tot);
        moe_convw_kernel<<<(tot + 255) / 256, 256, 0, stream>>>(W3, W3t, H_, H_, H_, tot);
        int tot4 = K_ * 16 * H_;
        moe_convw_kernel<<<(tot4 + 255) / 256, 256, 0, stream>>>(W4, W4t, H_, ACT_, 16, tot4);
    }

    moe_gate_kernel<<<B_ / GBR, 256, 0, stream>>>(x, Wg1, bg1, Wg2, bg2, gateW);

    const size_t smem = (size_t)(BM * H_ * 2 * 2)   // XT + ACT (bf16)
                      + (size_t)(WHALF * 2 * 2)     // two 64KB weight buffers
                      + (size_t)(BM * K_ * 4);      // gate tile
    moe_main_kernel<<<B_ / BM, 256, smem, stream>>>(
        x, W1t, W2t, W3t, W4t, b1, b2, b3, b4, gateW, out);
}
